// SATSolver_47459388621021
// MI455X (gfx1250) — matrix operand order repaired and checked by reading, not compiled, not run on hardware
//
#include <hip/hip_runtime.h>

#define NV 20000
#define NC 84000
#define NG 200
#define NE 252000
#define RNDS 8

typedef __attribute__((ext_vector_type(16))) _Float16 v16h;
typedef __attribute__((ext_vector_type(8)))  float    v8f;

__device__ __forceinline__ float sp(float x) {           // stable softplus
  return fmaxf(x, 0.0f) + log1pf(expf(-fabsf(x)));
}
__device__ __forceinline__ float sigm(float x) { return 1.0f / (1.0f + expf(-x)); }

// ---------------------------------------------------------------------------
__global__ void fill_f(float* p, float v, size_t n) {
  size_t i = (size_t)blockIdx.x * blockDim.x + threadIdx.x;
  if (i < n) p[i] = v;
}

// ---------------------------------------------------------------------------
// Pack f32 weight W[K,N] (row-major) into f16 WMMA B-fragments.
// Fragment layout (wave32, 32x16 f16 B): lane L holds column n0+(L&15),
// K values k0 + 8*(L>>4) + h for h<8, k0 + 16 + 8*(L>>4) + (h-8) for h>=8.  K zero-padded to multiple of 32.
__global__ void pack_w(const float* __restrict__ W, _Float16* __restrict__ P,
                       int K, int N) {
  int KT = (K + 31) >> 5, NT = N >> 4;
  int total = KT * NT * 512;
  int i = blockIdx.x * blockDim.x + threadIdx.x;
  if (i >= total) return;
  int h = i & 15, lane = (i >> 4) & 31, tile = i >> 9;
  int nt = tile % NT, kt = tile / NT;
  int n = nt * 16 + (lane & 15);
  int k = kt * 32 + ((h >> 3) << 4) + ((lane >> 4) << 3) + (h & 7);
  P[i] = (k < K) ? (_Float16)W[(size_t)k * N + n] : (_Float16)0.0f;
}

// ---------------------------------------------------------------------------
// C[M, NT*16] = act(A[M,K] @ Wpacked + bias).  One 16-row strip per wave:
// A fragment loaded once per K-chunk, NT WMMAs issued against it.
// K MUST be a multiple of 32 (callers pad buffers; no divergent guards).
// A fragment (16x32 f16): lane L holds row m0+(L&15); halves j / j+8 map to
// K = k0 + 8*(L>>4) + j  and  K = k0 + 16 + 8*(L>>4) + j (ISA 16-bit A layout).
template <int NT>
__global__ void gemm_wmma(const float* __restrict__ A, const _Float16* __restrict__ P,
                          const float* __restrict__ bias, float* __restrict__ C,
                          int M, int K, int relu) {
  constexpr int N = NT * 16;
  int KT = K >> 5;
  int MT = M >> 4;
  int lane = threadIdx.x & 31;
  int wave = blockIdx.x * (blockDim.x >> 5) + (threadIdx.x >> 5);
  int nw = gridDim.x * (blockDim.x >> 5);
  int r = lane & 15, hi = lane >> 4;
  for (int mt = wave; mt < MT; mt += nw) {
    const float* Arow = A + (size_t)(mt * 16 + r) * K + 8 * hi;
    v8f acc[NT];
#pragma unroll
    for (int t = 0; t < NT; ++t) acc[t] = {};
    for (int kt = 0; kt < KT; ++kt) {
      float4 a0 = *(const float4*)(Arow + 0);
      float4 a1 = *(const float4*)(Arow + 4);
      float4 a2 = *(const float4*)(Arow + 16);
      float4 a3 = *(const float4*)(Arow + 20);
      Arow += 32;
      v16h a;
      a[0] = (_Float16)a0.x;  a[1] = (_Float16)a0.y;
      a[2] = (_Float16)a0.z;  a[3] = (_Float16)a0.w;
      a[4] = (_Float16)a1.x;  a[5] = (_Float16)a1.y;
      a[6] = (_Float16)a1.z;  a[7] = (_Float16)a1.w;
      a[8]  = (_Float16)a2.x; a[9]  = (_Float16)a2.y;
      a[10] = (_Float16)a2.z; a[11] = (_Float16)a2.w;
      a[12] = (_Float16)a3.x; a[13] = (_Float16)a3.y;
      a[14] = (_Float16)a3.z; a[15] = (_Float16)a3.w;
      const _Float16* Pk = P + ((size_t)kt * NT * 512) + (size_t)lane * 16;
#pragma unroll
      for (int t = 0; t < NT; ++t) {
        v16h b = *(const v16h*)(Pk + (size_t)t * 512);
        acc[t] = __builtin_amdgcn_wmma_f32_16x16x32_f16(false, a, false, b,
                                                        (short)0, acc[t], false, false);
      }
    }
#pragma unroll
    for (int t = 0; t < NT; ++t) {
      int col = t * 16 + r;
      float bv = bias ? bias[col] : 0.0f;
      float* Cb = C + (size_t)(mt * 16 + hi * 8) * N + col;
#pragma unroll
      for (int i = 0; i < 8; ++i) {
        float v = acc[t][i] + bv;
        if (relu) v = fmaxf(v, 0.0f);
        Cb[(size_t)i * N] = v;
      }
    }
  }
}

// ---------------------------------------------------------------------------
// Precompute: degrees, masks, norms
__global__ void edge_pre(const int* __restrict__ elit, const int* __restrict__ eclause,
                         const float* __restrict__ wsig, const float* __restrict__ wsp,
                         float* litdeg, float* varsum) {
  int e = blockIdx.x * blockDim.x + threadIdx.x;
  if (e >= NE) return;
  int c = eclause[e]; int l = elit[e];
  float w = wsig[c];
  atomicAdd(&litdeg[l], w);
  atomicAdd(&varsum[l], w * wsp[c]);
}

__global__ void var_pre(const float* litdeg, const float* varsum, const int* vgid,
                        float* vmask, float* gvm, float* degw, float* vdegw) {
  int v = blockIdx.x * blockDim.x + threadIdx.x;
  if (v >= NV) return;
  float vm = 1.0f - expf(-(varsum[v] + varsum[v + NV]));
  vmask[v] = vm;
  atomicAdd(&gvm[vgid[v]], vm);
  float l0 = litdeg[v], l1 = litdeg[v + NV];
  degw[v]      = rsqrtf(fmaxf(l0, 1.0f));
  degw[v + NV] = rsqrtf(fmaxf(l1, 1.0f));
  vdegw[v] = 4.0f * rsqrtf(fmaxf(l0 + l1, 1.0f));
}

__global__ void clause_pre(const float* wsig, const int* cgid, float* gws) {
  int c = blockIdx.x * blockDim.x + threadIdx.x;
  if (c >= NC) return;
  atomicAdd(&gws[cgid[c]], wsig[c]);
}

__global__ void norm_fin(const float* vmask, const int* vgid, const float* gvm, float* vgn,
                         const float* wsig, const int* cgid, const float* gws, float* cgn) {
  int i = blockIdx.x * blockDim.x + threadIdx.x;
  if (i < NV) { float s = gvm[vgid[i]]; vgn[i] = vmask[i] * (s > 0.0f ? 1.0f / s : 0.0f); }
  if (i < NC) { float s = gws[cgid[i]]; cgn[i] = wsig[i] * (s > 0.0f ? 1.0f / s : 0.0f); }
}

// ---------------------------------------------------------------------------
// Per-round kernels
// v1 padded to stride 96 (cols 68..95 = 0) so GEMM needs no K guards.
__global__ void build_v1(const float* vars, const float* noise_t, float* v1) {
  int i = blockIdx.x * blockDim.x + threadIdx.x;
  if (i >= NV * 96) return;
  int v = i / 96, k = i - v * 96;
  float val;
  if (k < 64)      val = vars[(size_t)v * 64 + k];
  else if (k < 68) val = noise_t[(size_t)v * 4 + (k - 64)];
  else             val = 0.0f;
  v1[i] = val;
}

__global__ void clause_fwd(const float* __restrict__ query, const int* __restrict__ elit,
                           const float* __restrict__ wsig, const float* __restrict__ cs_in,
                           float* closs, float* cunit) {
  int i = blockIdx.x * blockDim.x + threadIdx.x;
  if (i >= NC * 64) return;
  int c = i >> 6, f = i & 63;
  float w = wsig[c];
  float cval = 0.0f;
#pragma unroll
  for (int j = 0; j < 3; ++j) {
    int l = elit[3 * c + j];
    float q = (l < NV) ? query[(size_t)l * 64 + f] : -query[(size_t)(l - NV) * 64 + f];
    cval += sp(q) * w;
  }
  float cl = expf(-cval);
  closs[i] = cl;
  cunit[(size_t)c * 128 + f] = cs_in[i];
  cunit[(size_t)c * 128 + 64 + f] = 4.0f * cl;
}

__global__ void grad_scatter(const float* __restrict__ closs, const int* __restrict__ elit,
                             const int* __restrict__ eclause, const float* __restrict__ wsig,
                             float* gacc) {
  int i = blockIdx.x * blockDim.x + threadIdx.x;
  if (i >= NE * 64) return;
  int e = i >> 6, f = i & 63;
  int c = eclause[e]; int l = elit[e];
  atomicAdd(&gacc[(size_t)l * 64 + f], -closs[(size_t)c * 64 + f] * wsig[c]);
}

__global__ void unit_p1(const float* query, const float* gacc, const float* vdegw,
                        const float* vars, float* unit) {
  int i = blockIdx.x * blockDim.x + threadIdx.x;
  if (i >= NV * 64) return;
  int v = i >> 6, f = i & 63;
  float q = query[i];
  float s = sigm(q);
  float g = gacc[i] * s - gacc[(size_t)(v + NV) * 64 + f] * (1.0f - s);
  unit[(size_t)v * 256 + f] = g * vdegw[v];
  unit[(size_t)v * 256 + 64 + f] = vars[i];
}

__global__ void pn_mean(const float* x, int stride, int offset, const float* wn,
                        const int* seg, float* mean, int M) {
  int i = blockIdx.x * blockDim.x + threadIdx.x;
  if (i >= M * 64) return;
  int c = i >> 6, f = i & 63;
  atomicAdd(&mean[(size_t)seg[c] * 64 + f], x[(size_t)c * stride + offset + f] * wn[c]);
}

__global__ void pn_var(const float* x, int stride, int offset, const float* wn,
                       const int* seg, const float* mean, float* var, int M) {
  int c = blockIdx.x * blockDim.x + threadIdx.x;
  if (c >= M) return;
  const float* mrow = &mean[(size_t)seg[c] * 64];
  float s = 0.0f;
  for (int f = 0; f < 64; ++f) {
    float d = x[(size_t)c * stride + offset + f] - mrow[f];
    s += d * d;
  }
  atomicAdd(&var[seg[c]], (s * (1.0f / 64.0f)) * wn[c]);
}

__global__ void pn_apply_clause(const float* cdata, const float* mean, const float* var,
                                const int* seg, const float* cs_in, float* cs_out,
                                const int* done_r) {
  int i = blockIdx.x * blockDim.x + threadIdx.x;
  if (i >= NC * 64) return;
  int c = i >> 6, f = i & 63;
  int g = seg[c];
  float xc = cdata[(size_t)c * 128 + 64 + f] - mean[(size_t)g * 64 + f];
  float ncs = xc * rsqrtf(var[g] + 1e-6f) * 0.25f + 0.1f * cs_in[i];
  cs_out[i] = (*done_r) ? cs_in[i] : ncs;
}

__global__ void vloss_scatter(const float* cdata, const int* elit, const int* eclause,
                              const float* wsig, float* vlacc) {
  int i = blockIdx.x * blockDim.x + threadIdx.x;
  if (i >= NE * 64) return;
  int e = i >> 6, f = i & 63;
  int c = eclause[e]; int l = elit[e];
  atomicAdd(&vlacc[(size_t)l * 64 + f], cdata[(size_t)c * 128 + f] * wsig[c]);
}

__global__ void unit_p2(const float* vlacc, const float* degw, float* unit) {
  int i = blockIdx.x * blockDim.x + threadIdx.x;
  if (i >= NV * 64) return;
  int v = i >> 6, f = i & 63;
  unit[(size_t)v * 256 + 128 + f] = vlacc[i] * degw[v];
  unit[(size_t)v * 256 + 192 + f] = vlacc[(size_t)(v + NV) * 64 + f] * degw[v + NV];
}

__global__ void pn_apply_var(const float* uout, const float* mean, const float* var,
                             const int* seg, const float* vars_in, float* nvbuf,
                             float* vars_out, const int* done_r) {
  int i = blockIdx.x * blockDim.x + threadIdx.x;
  if (i >= NV * 64) return;
  int v = i >> 6, f = i & 63;
  int g = seg[v];
  float xc = uout[i] - mean[(size_t)g * 64 + f];
  float nvl = xc * rsqrtf(var[g] + 1e-6f) * 0.25f + 0.1f * vars_in[i];
  nvbuf[i] = nvl;
  vars_out[i] = (*done_r) ? vars_in[i] : nvl;
}

__global__ void logits_k(const float* oh, const float* ow2, const float* ob2,
                         float* logits, float* lastlog, const int* done_r) {
  int v = blockIdx.x * blockDim.x + threadIdx.x;
  if (v >= NV) return;
  float s = ob2[0];
  for (int f = 0; f < 64; ++f) s += oh[(size_t)v * 64 + f] * ow2[f];
  logits[v] = s;
  if (!*done_r) lastlog[v] = s;
}

__global__ void clause_loss_sat(const float* logits, const int* elit, const float* wsig,
                                const int* cgid, float* pg, int* satflag) {
  int c = blockIdx.x * blockDim.x + threadIdx.x;
  if (c >= NC) return;
  float w = wsig[c];
  float cval = 0.0f, cs = 0.0f;
#pragma unroll
  for (int j = 0; j < 3; ++j) {
    int l = elit[3 * c + j];
    int v = (l < NV) ? l : l - NV;
    float lg = logits[v];
    cval += sp((l < NV) ? lg : -lg) * w;
    float a = roundf(sigm(lg));
    cs += ((l < NV) ? a : 1.0f - a) * w;
  }
  float cv = expf(-cval);
  float pc = cv * -logf(1.0f - cv + 1e-10f);
  atomicAdd(&pg[cgid[c]], pc * w);
  if (cs < 1.0f) *satflag = 0;
}

__global__ void round_init(float* pg, float* cmean, float* cvar, float* vmean,
                           float* vvar, int* satflag) {
  int i = blockIdx.x * blockDim.x + threadIdx.x;
  if (i == 0) *satflag = 1;
  if (i < NG) { pg[i] = 0.0f; cvar[i] = 0.0f; vvar[i] = 0.0f; }
  if (i < NG * 64) { cmean[i] = 0.0f; vmean[i] = 0.0f; }
}

__global__ void round_fin(const float* pg, float* losss, int* done, int* step,
                          const int* satflag, int r) {
  int g = blockIdx.x * blockDim.x + threadIdx.x;
  int d = done[r];
  if (g < NG && !d) losss[g] += sqrtf(pg[g] + 1e-6f) - sqrtf(1e-6f);
  if (g == 0) {
    done[r + 1] = d | *satflag;
    if (!d) *step = *step + 1;
  }
}

__global__ void write_out(const float* lastlog, const float* losss, const int* step,
                          float* out) {
  int i = blockIdx.x * blockDim.x + threadIdx.x;
  if (i > NV + NG) return;
  if (i < NV) out[i] = lastlog[i];
  else if (i < NV + NG) out[i] = losss[i - NV] * (1.0f / (float)RNDS);
  else out[i] = (float)(*step);
}

// ---------------------------------------------------------------------------
extern "C" void kernel_launch(void* const* d_in, const int* in_sizes, int n_in,
                              void* d_out, int out_size, void* d_ws, size_t ws_size,
                              hipStream_t stream) {
  (void)in_sizes; (void)n_in; (void)out_size; (void)ws_size;
  const float* wsig   = (const float*)d_in[0];
  const float* wsp    = (const float*)d_in[1];
  const float* noise  = (const float*)d_in[2];
  const int*   elit   = (const int*)d_in[3];
  const int*   eclause= (const int*)d_in[4];
  const int*   cgid   = (const int*)d_in[5];
  const int*   vgid   = (const int*)d_in[6];
  const float* q_w1 = (const float*)d_in[7],  *q_b1 = (const float*)d_in[8];
  const float* q_w2 = (const float*)d_in[9],  *q_b2 = (const float*)d_in[10];
  const float* c_w1 = (const float*)d_in[11], *c_b1 = (const float*)d_in[12];
  const float* c_w2 = (const float*)d_in[13], *c_b2 = (const float*)d_in[14];
  const float* u_w1 = (const float*)d_in[15], *u_b1 = (const float*)d_in[16];
  const float* u_w2 = (const float*)d_in[17], *u_b2 = (const float*)d_in[18];
  const float* u_w3 = (const float*)d_in[19], *u_b3 = (const float*)d_in[20];
  const float* o_w1 = (const float*)d_in[21], *o_b1 = (const float*)d_in[22];
  const float* o_w2 = (const float*)d_in[23], *o_b2 = (const float*)d_in[24];
  float* out = (float*)d_out;
  float* W = (float*)d_ws;

  // ---- workspace layout (float units, 64-aligned) ----
  size_t off = 0;
  auto alloc = [&](size_t n) { size_t o = off; off += (n + 63) & ~(size_t)63; return o; };
  size_t vars0 = alloc((size_t)NV * 64), vars1 = alloc((size_t)NV * 64);
  size_t cs0 = alloc((size_t)NC * 64),   cs1 = alloc((size_t)NC * 64);
  size_t lastlog = alloc(NV), losss = alloc(NG);
  size_t vgn = alloc(NV), cgn = alloc(NC);
  size_t degw = alloc(2 * NV), vdegw = alloc(NV);
  size_t litdeg = alloc(2 * NV), varsum = alloc(2 * NV), vmask = alloc(NV);
  size_t gvm = alloc(NG), gws = alloc(NG), pg = alloc(NG);
  size_t cmean = alloc(NG * 64), cvar = alloc(NG);
  size_t vmean = alloc(NG * 64), vvar = alloc(NG);
  size_t logits = alloc(NV);
  size_t qh    = alloc((size_t)NV * 64);   // also: nvbuf (reused)
  size_t query = alloc((size_t)NV * 64);   // also: uout  (reused)
  size_t closs = alloc((size_t)NC * 64);   // also: oh    (reused)
  size_t cunit = alloc((size_t)NC * 128);
  size_t chbuf = alloc((size_t)NC * 128);  // ch; then gacc/vlacc/uh1/uh2 carved here
  size_t cdata = alloc((size_t)NC * 128);
  size_t unit  = alloc((size_t)NV * 256);  // v1 (stride 96) aliases its start
  size_t intoff = alloc(64);
  size_t halfoff = alloc(104448 / 2 + 64); // packed f16 weights (104448 halves)

  float* gacc = W + chbuf;                          // [2*NV,64]
  float* vlacc = W + chbuf + (size_t)2 * NV * 64;   // [2*NV,64]
  float* uh1 = W + chbuf + (size_t)4 * NV * 64;     // [NV,128]
  float* uh2 = uh1 + (size_t)NV * 128;              // [NV,128]
  float* uout = W + query;                          // [NV,64]
  float* nvbuf = W + qh;                            // [NV,64]
  float* oh = W + closs;                            // [NV,64]
  int* done = (int*)(W + intoff);                   // done[0..8]
  int* step = done + 9;
  int* satflag = done + 10;

  _Float16* H = (_Float16*)(W + halfoff);
  _Float16* Pq1 = H;            // K=68->96 (KT3), N=64 (NT4): 6144
  _Float16* Pq2 = Pq1 + 6144;   // 4096
  _Float16* Pc1 = Pq2 + 4096;   // 16384
  _Float16* Pc2 = Pc1 + 16384;  // 16384
  _Float16* Pu1 = Pc2 + 16384;  // 32768
  _Float16* Pu2 = Pu1 + 32768;  // 16384
  _Float16* Pu3 = Pu2 + 16384;  // 8192
  _Float16* Po1 = Pu3 + 8192;   // 4096

#define LB(n) ((int)(((n) + 255) / 256)), 256, 0, stream
  auto gemm = [&](const float* A, const _Float16* P, const float* b, float* C,
                  int M, int K, int N, int relu) {
    int MT = M >> 4;
    int blocks = (MT + 7) / 8;
    if (N == 64) gemm_wmma<4><<<blocks, 256, 0, stream>>>(A, P, b, C, M, K, relu);
    else         gemm_wmma<8><<<blocks, 256, 0, stream>>>(A, P, b, C, M, K, relu);
  };

  // ---- init ----
  hipMemsetAsync(W + litdeg, 0, sizeof(float) * 2 * NV, stream);
  hipMemsetAsync(W + varsum, 0, sizeof(float) * 2 * NV, stream);
  hipMemsetAsync(W + gvm, 0, sizeof(float) * NG, stream);
  hipMemsetAsync(W + gws, 0, sizeof(float) * NG, stream);
  hipMemsetAsync(W + lastlog, 0, sizeof(float) * NV, stream);
  hipMemsetAsync(W + losss, 0, sizeof(float) * NG, stream);
  hipMemsetAsync(done, 0, sizeof(int) * 11, stream);
  fill_f<<<LB((size_t)NV * 64)>>>(W + vars0, 1.0f, (size_t)NV * 64);
  fill_f<<<LB((size_t)NC * 64)>>>(W + cs0, 1.0f, (size_t)NC * 64);

  // pack weights to f16 fragments
  pack_w<<<LB(3 * 4 * 512)>>>(q_w1, Pq1, 68, 64);
  pack_w<<<LB(2 * 4 * 512)>>>(q_w2, Pq2, 64, 64);
  pack_w<<<LB(4 * 8 * 512)>>>(c_w1, Pc1, 128, 128);
  pack_w<<<LB(4 * 8 * 512)>>>(c_w2, Pc2, 128, 128);
  pack_w<<<LB(8 * 8 * 512)>>>(u_w1, Pu1, 256, 128);
  pack_w<<<LB(4 * 8 * 512)>>>(u_w2, Pu2, 128, 128);
  pack_w<<<LB(4 * 4 * 512)>>>(u_w3, Pu3, 128, 64);
  pack_w<<<LB(2 * 4 * 512)>>>(o_w1, Po1, 64, 64);

  // precompute
  edge_pre<<<LB(NE)>>>(elit, eclause, wsig, wsp, W + litdeg, W + varsum);
  var_pre<<<LB(NV)>>>(W + litdeg, W + varsum, vgid, W + vmask, W + gvm, W + degw, W + vdegw);
  clause_pre<<<LB(NC)>>>(wsig, cgid, W + gws);
  norm_fin<<<LB(NC)>>>(W + vmask, vgid, W + gvm, W + vgn, wsig, cgid, W + gws, W + cgn);

  // ---- rounds ----
  for (int r = 0; r < RNDS; ++r) {
    float* vin = W + ((r & 1) ? vars1 : vars0);
    float* vout = W + ((r & 1) ? vars0 : vars1);
    float* cin = W + ((r & 1) ? cs1 : cs0);
    float* cout = W + ((r & 1) ? cs0 : cs1);
    const float* noise_t = noise + (size_t)r * NV * 4;
    const int* done_r = done + r;

    round_init<<<LB(NG * 64)>>>(W + pg, W + cmean, W + cvar, W + vmean, W + vvar, satflag);

    // q-MLP  (v1 padded to K=96 in unit region)
    build_v1<<<LB(NV * 96)>>>(vin, noise_t, W + unit);
    gemm(W + unit, Pq1, q_b1, W + qh, NV, 96, 64, 1);
    gemm(W + qh, Pq2, q_b2, W + query, NV, 64, 64, 0);

    // clause loss + clause MLP
    clause_fwd<<<LB((size_t)NC * 64)>>>(W + query, elit, wsig, cin, W + closs, W + cunit);
    gemm(W + cunit, Pc1, c_b1, W + chbuf, NC, 128, 128, 1);
    gemm(W + chbuf, Pc2, c_b2, W + cdata, NC, 128, 128, 0);

    // ch region now free -> gacc + vlacc
    hipMemsetAsync(W + chbuf, 0, sizeof(float) * (size_t)4 * NV * 64, stream);
    grad_scatter<<<LB((size_t)NE * 64)>>>(W + closs, elit, eclause, wsig, gacc);
    unit_p1<<<LB((size_t)NV * 64)>>>(W + query, gacc, W + vdegw, vin, W + unit);

    // clause pair-norm -> new clause state
    pn_mean<<<LB((size_t)NC * 64)>>>(W + cdata, 128, 64, W + cgn, cgid, W + cmean, NC);
    pn_var<<<LB(NC)>>>(W + cdata, 128, 64, W + cgn, cgid, W + cmean, W + cvar, NC);
    pn_apply_clause<<<LB((size_t)NC * 64)>>>(W + cdata, W + cmean, W + cvar, cgid,
                                             cin, cout, done_r);

    // variables_loss scatter + unit completion
    vloss_scatter<<<LB((size_t)NE * 64)>>>(W + cdata, elit, eclause, wsig, vlacc);
    unit_p2<<<LB((size_t)NV * 64)>>>(vlacc, W + degw, W + unit);

    // u-MLP
    gemm(W + unit, Pu1, u_b1, uh1, NV, 256, 128, 1);
    gemm(uh1, Pu2, u_b2, uh2, NV, 128, 128, 1);
    gemm(uh2, Pu3, u_b3, uout, NV, 128, 64, 0);

    // variable pair-norm -> new variables
    pn_mean<<<LB((size_t)NV * 64)>>>(uout, 64, 0, W + vgn, vgid, W + vmean, NV);
    pn_var<<<LB(NV)>>>(uout, 64, 0, W + vgn, vgid, W + vmean, W + vvar, NV);
    pn_apply_var<<<LB((size_t)NV * 64)>>>(uout, W + vmean, W + vvar, vgid,
                                          vin, nvbuf, vout, done_r);

    // o-MLP + logits
    gemm(nvbuf, Po1, o_b1, oh, NV, 64, 64, 1);
    logits_k<<<LB(NV)>>>(oh, o_w2, o_b2, W + logits, W + lastlog, done_r);

    // per-clause mixed loss + sat check
    clause_loss_sat<<<LB(NC)>>>(W + logits, elit, wsig, cgid, W + pg, satflag);
    round_fin<<<LB(NG)>>>(W + pg, W + losss, done, step, satflag, r);
  }

  write_out<<<LB(NV + NG + 1)>>>(W + lastlog, W + losss, step, out);
#undef LB
}
